// Net_16149077033178
// MI455X (gfx1250) — compile-verified, matrix operand order checked by reading, not run on hardware
//
#include <hip/hip_runtime.h>
#include <hip/hip_bf16.h>
#include <math.h>

#define N_NODES 100000
#define N_EDGES 3200000
#define F_IN    1433
#define KPAD    1440            // 45 * 32
#define NCHUNK  45
#define KTAIL   1408            // start of last chunk (ch = 44)
#define NTILES  6250            // 100000 / 16
#define NCAT    24              // [xw_k0(8) | xw_k1(8) | h(8)] per node, stride 24
#define WPACK_DW (NCHUNK * 2 * 32 * 8)     // 23040 dwords
#define WLDS_BYTES (WPACK_DW * 4)          // 92160 B of LDS (CDNA5: 320KB/WGP)

typedef __attribute__((ext_vector_type(16))) __bf16 v16bf;
typedef __attribute__((ext_vector_type(8)))  float  v8f;

// ---------- float <-> orderable-uint key (for atomic segment-max) ----------
__device__ __forceinline__ unsigned f2key(float f) {
    unsigned b = __float_as_uint(f);
    return (b & 0x80000000u) ? ~b : (b | 0x80000000u);
}
__device__ __forceinline__ float key2f(unsigned k) {
    unsigned b = (k & 0x80000000u) ? (k & 0x7fffffffu) : ~k;
    return __uint_as_float(b);
}
#define NEG_INF_KEY 0x007FFFFFu   // f2key(-inf)

__device__ __forceinline__ unsigned pk_bf16(float lo, float hi) {
    union { __bf16 h; unsigned short u; } a, b;
    a.h = (__bf16)lo; b.h = (__bf16)hi;
    return (unsigned)a.u | ((unsigned)b.u << 16);
}

// =====================================================================
// Kernel 0a: pack W = [w1_k0 | w1_k1 | root1] (1433x24 -> 1440x32, bf16)
// directly into the per-lane WMMA B-fragment layout:
//   dword index = ((chunk*2 + ntile)*32 + lane)*8 + v
//   lane: n = ntile*16 + (lane&15), h = lane>>4
//   v:    k = chunk*32 + (v<4?0:16) + h*8 + 2*(v&3)  (pair k, k+1)
// =====================================================================
__global__ __launch_bounds__(256) void pack_w1_kernel(
        const float* __restrict__ w1, const float* __restrict__ root1,
        unsigned* __restrict__ wpack) {
    int tid = blockIdx.x * 256 + threadIdx.x;
    if (tid >= WPACK_DW) return;
    int v    = tid & 7;
    int lane = (tid >> 3) & 31;
    int nt   = (tid >> 8) & 1;
    int ch   = tid >> 9;
    int n  = nt * 16 + (lane & 15);
    int hh = lane >> 4;
    int k  = ch * 32 + ((v < 4) ? 0 : 16) + hh * 8 + 2 * (v & 3);
    float f[2] = {0.f, 0.f};
#pragma unroll
    for (int s = 0; s < 2; ++s) {
        int kk = k + s;
        if (kk < F_IN && n < 24) {
            if (n < 16) f[s] = w1[((size_t)(n >> 3) * F_IN + kk) * 8 + (n & 7)];
            else        f[s] = root1[(size_t)kk * 8 + (n - 16)];
        }
    }
    wpack[tid] = pk_bf16(f[0], f[1]);
}

// =====================================================================
// Kernel 0b: zero-padded copy of x's last 32 columns (k = 1408..1439)
// so the GEMM K-tail needs no per-element bounds checks.
// =====================================================================
__global__ __launch_bounds__(256) void pack_xtail_kernel(
        const float* __restrict__ x, float* __restrict__ xtail) {
    int tid = blockIdx.x * 256 + threadIdx.x;
    if (tid >= N_NODES * 32) return;
    int n = tid >> 5, c = tid & 31;
    int k = KTAIL + c;
    xtail[tid] = (k < F_IN) ? x[(size_t)n * F_IN + k] : 0.f;
}

// =====================================================================
// Kernel 1: fused layer-1 GEMM via WMMA bf16.
//   nodecat[n, 0:24] = x[n,:] @ [w1_k0 | w1_k1 | root1]
// One wave per 16-node tile; 8 waves per block; 45 k-chunks of 32.
// Packed W is staged once per block into LDS (ds_load_b128 in the loop).
// =====================================================================
__global__ __launch_bounds__(256) void gemm1_wmma_kernel(
        const float* __restrict__ x, const float* __restrict__ xtail,
        const unsigned* __restrict__ wpack, float* __restrict__ nodecat) {
    extern __shared__ unsigned wlds[];

    // cooperative, coalesced copy of packed W into LDS (5760 x b128)
    {
        const uint4* src = (const uint4*)wpack;
        uint4*       dst = (uint4*)wlds;
        for (int i = threadIdx.x; i < WPACK_DW / 4; i += 256)
            dst[i] = src[i];
    }
    __syncthreads();

    int lane = threadIdx.x & 31;
    int wave = threadIdx.x >> 5;
    int t = blockIdx.x * 8 + wave;            // tile index
    if (t >= NTILES) return;                  // after the barrier: safe

    int m  = lane & 15;                       // A-fragment row this lane covers
    int hh = lane >> 4;                       // lane half selects K sub-block
    const float* xrow  = x     + (size_t)(t * 16 + m) * F_IN;
    const float* xtrow = xtail + (size_t)(t * 16 + m) * 32;

    v8f c0 = {};                              // cols 0..15
    v8f c1 = {};                              // cols 16..31 (we keep 16..23)

    // ---- 44 full chunks: no bounds checks (max k+1 = 1407 < 1433) ----
    for (int ch = 0; ch < NCHUNK - 1; ++ch) {
        int kbase = ch * 32;
        __builtin_prefetch(xrow + kbase + 32, 0, 3);   // global_prefetch_b8

        union { unsigned u[8]; v16bf v; } ua;
#pragma unroll
        for (int v = 0; v < 8; ++v) {
            int k = kbase + ((v < 4) ? 0 : 16) + hh * 8 + 2 * (v & 3);
            ua.u[v] = pk_bf16(xrow[k], xrow[k + 1]);
        }

        const uint4* b0p = (const uint4*)(wlds + ((ch * 2 + 0) * 32 + lane) * 8);
        const uint4* b1p = (const uint4*)(wlds + ((ch * 2 + 1) * 32 + lane) * 8);
        union { uint4 q[2]; v16bf v; } ub0, ub1;
        ub0.q[0] = b0p[0]; ub0.q[1] = b0p[1];
        ub1.q[0] = b1p[0]; ub1.q[1] = b1p[1];

        c0 = __builtin_amdgcn_wmma_f32_16x16x32_bf16(false, ua.v, false, ub0.v,
                                                     (short)0, c0, false, false);
        c1 = __builtin_amdgcn_wmma_f32_16x16x32_bf16(false, ua.v, false, ub1.v,
                                                     (short)0, c1, false, false);
    }

    // ---- final chunk from the zero-padded tail buffer: no checks ----
    {
        const int ch = NCHUNK - 1;
        union { unsigned u[8]; v16bf v; } ua;
#pragma unroll
        for (int v = 0; v < 8; ++v) {
            int k = ((v < 4) ? 0 : 16) + hh * 8 + 2 * (v & 3);
            ua.u[v] = pk_bf16(xtrow[k], xtrow[k + 1]);
        }
        const uint4* b0p = (const uint4*)(wlds + ((ch * 2 + 0) * 32 + lane) * 8);
        const uint4* b1p = (const uint4*)(wlds + ((ch * 2 + 1) * 32 + lane) * 8);
        union { uint4 q[2]; v16bf v; } ub0, ub1;
        ub0.q[0] = b0p[0]; ub0.q[1] = b0p[1];
        ub1.q[0] = b1p[0]; ub1.q[1] = b1p[1];

        c0 = __builtin_amdgcn_wmma_f32_16x16x32_bf16(false, ua.v, false, ub0.v,
                                                     (short)0, c0, false, false);
        c1 = __builtin_amdgcn_wmma_f32_16x16x32_bf16(false, ua.v, false, ub1.v,
                                                     (short)0, c1, false, false);
    }

    // ---- epilogue: D layout = VGPR r -> row (r + 8*hh), lane%16 -> col ----
    int n0 = lane & 15;
#pragma unroll
    for (int r = 0; r < 8; ++r) {
        int node = t * 16 + r + 8 * hh;
        nodecat[(size_t)node * NCAT + n0] = c0[r];
        if (n0 < 8)
            nodecat[(size_t)node * NCAT + 16 + n0] = c1[r];
    }
}

// =====================================================================
// Per-node: hdot[n] = h[n] . att[OD:], and init segment-reduce buffers
// =====================================================================
template <int OD>
__global__ __launch_bounds__(256) void node_hdot_init_kernel(
        const float* __restrict__ nodecat, const float* __restrict__ att,
        float* __restrict__ hdot, unsigned* __restrict__ smaxkey,
        float* __restrict__ denom, float* __restrict__ agg) {
    int n = blockIdx.x * 256 + threadIdx.x;
    if (n >= N_NODES) return;
    const float* row = nodecat + (size_t)n * NCAT + 16;
    float s = 0.f;
#pragma unroll
    for (int j = 0; j < OD; ++j) s += row[j] * att[OD + j];
    hdot[n] = s;
    smaxkey[n] = NEG_INF_KEY;
    denom[n] = 0.f;
#pragma unroll
    for (int j = 0; j < 8; ++j) agg[(size_t)n * 8 + j] = 0.f;
}

// =====================================================================
// Edge pass A: score + leaky-relu + atomic segment-max
// =====================================================================
template <int OD>
__global__ __launch_bounds__(256) void edge_score_max_kernel(
        const long long* __restrict__ ei, const float* __restrict__ pseudo,
        const float* __restrict__ nodecat, const float* __restrict__ hdot,
        const float* __restrict__ att, float* __restrict__ escore,
        unsigned* __restrict__ smaxkey) {
    int e = blockIdx.x * 256 + threadIdx.x;
    if (e >= N_EDGES) return;
    int s = (int)ei[e];
    int d = (int)ei[(size_t)N_EDGES + e];
    float u = pseudo[e];
    const float* row = nodecat + (size_t)s * NCAT;
    float sc = 0.f;
#pragma unroll
    for (int j = 0; j < OD; ++j) {
        float mj = row[j] * (1.f - u) + row[8 + j] * u;
        sc += mj * att[j];
    }
    sc += hdot[d];
    sc = (sc > 0.f) ? sc : 0.2f * sc;          // leaky relu
    escore[e] = sc;
    atomicMax(&smaxkey[d], f2key(sc));
}

// =====================================================================
// Edge pass B: e = exp(score - smax[dst]); atomic segment-sum -> denom
// =====================================================================
__global__ __launch_bounds__(256) void edge_expsum_kernel(
        const long long* __restrict__ ei, float* __restrict__ escore,
        const unsigned* __restrict__ smaxkey, float* __restrict__ denom) {
    int e = blockIdx.x * 256 + threadIdx.x;
    if (e >= N_EDGES) return;
    int d = (int)ei[(size_t)N_EDGES + e];
    float ex = __expf(escore[e] - key2f(smaxkey[d]));
    escore[e] = ex;
    atomicAdd(&denom[d], ex);
}

// =====================================================================
// Edge pass C: alpha = e/denom[dst]; agg[dst] += alpha * m  (recompute m)
// =====================================================================
template <int OD>
__global__ __launch_bounds__(256) void edge_agg_kernel(
        const long long* __restrict__ ei, const float* __restrict__ pseudo,
        const float* __restrict__ nodecat, const float* __restrict__ escore,
        const float* __restrict__ denom, float* __restrict__ agg) {
    int e = blockIdx.x * 256 + threadIdx.x;
    if (e >= N_EDGES) return;
    int s = (int)ei[e];
    int d = (int)ei[(size_t)N_EDGES + e];
    float u = pseudo[e];
    float alpha = escore[e] / (denom[d] + 1e-16f);
    const float* row = nodecat + (size_t)s * NCAT;
#pragma unroll
    for (int j = 0; j < OD; ++j) {
        float mj = row[j] * (1.f - u) + row[8 + j] * u;
        atomicAdd(&agg[(size_t)d * 8 + j], alpha * mj);
    }
}

// =====================================================================
// Layer-1 output: elu(agg + h + bias1)  -> hl2[100000, 8]
// =====================================================================
__global__ __launch_bounds__(256) void node_out1_kernel(
        const float* __restrict__ nodecat, const float* __restrict__ agg,
        const float* __restrict__ bias1, float* __restrict__ hl2) {
    int n = blockIdx.x * 256 + threadIdx.x;
    if (n >= N_NODES) return;
#pragma unroll
    for (int j = 0; j < 8; ++j) {
        float v = agg[(size_t)n * 8 + j] + nodecat[(size_t)n * NCAT + 16 + j] + bias1[j];
        hl2[(size_t)n * 8 + j] = (v > 0.f) ? v : (expf(v) - 1.f);
    }
}

// =====================================================================
// Layer-2 tiny GEMM (K=8): nodecat[n] = hl2[n,:] @ [w2_k0|w2_k1|root2]
// =====================================================================
__global__ __launch_bounds__(256) void node_l2_kernel(
        const float* __restrict__ hl2, const float* __restrict__ w2,
        const float* __restrict__ root2, float* __restrict__ nodecat) {
    int n = blockIdx.x * 256 + threadIdx.x;
    if (n >= N_NODES) return;
    float h[8];
#pragma unroll
    for (int i = 0; i < 8; ++i) h[i] = hl2[(size_t)n * 8 + i];
#pragma unroll
    for (int o = 0; o < 7; ++o) {
        float a0 = 0.f, a1 = 0.f, hh = 0.f;
#pragma unroll
        for (int i = 0; i < 8; ++i) {
            a0 += h[i] * w2[(size_t)(0 * 8 + i) * 7 + o];
            a1 += h[i] * w2[(size_t)(1 * 8 + i) * 7 + o];
            hh += h[i] * root2[(size_t)i * 7 + o];
        }
        nodecat[(size_t)n * NCAT + o]      = a0;
        nodecat[(size_t)n * NCAT + 8 + o]  = a1;
        nodecat[(size_t)n * NCAT + 16 + o] = hh;
    }
}

// =====================================================================
// Final: out = log_softmax(agg + h2 + bias2)
// =====================================================================
__global__ __launch_bounds__(256) void node_final_kernel(
        const float* __restrict__ nodecat, const float* __restrict__ agg,
        const float* __restrict__ bias2, float* __restrict__ out) {
    int n = blockIdx.x * 256 + threadIdx.x;
    if (n >= N_NODES) return;
    float v[7];
    float mx = -INFINITY;
#pragma unroll
    for (int o = 0; o < 7; ++o) {
        v[o] = agg[(size_t)n * 8 + o] + nodecat[(size_t)n * NCAT + 16 + o] + bias2[o];
        mx = fmaxf(mx, v[o]);
    }
    float s = 0.f;
#pragma unroll
    for (int o = 0; o < 7; ++o) s += expf(v[o] - mx);
    float ls = logf(s);
#pragma unroll
    for (int o = 0; o < 7; ++o) out[(size_t)n * 7 + o] = v[o] - mx - ls;
}

// =====================================================================
extern "C" void kernel_launch(void* const* d_in, const int* in_sizes, int n_in,
                              void* d_out, int out_size, void* d_ws, size_t ws_size,
                              hipStream_t stream) {
    const float*     x      = (const float*)d_in[0];
    const long long* ei     = (const long long*)d_in[1];
    const float*     pseudo = (const float*)d_in[2];
    const float*     w1     = (const float*)d_in[3];
    const float*     root1  = (const float*)d_in[4];
    const float*     att1   = (const float*)d_in[5];
    const float*     bias1  = (const float*)d_in[6];
    const float*     w2     = (const float*)d_in[7];
    const float*     root2  = (const float*)d_in[8];
    const float*     att2   = (const float*)d_in[9];
    const float*     bias2  = (const float*)d_in[10];
    float* out = (float*)d_out;

    // workspace carve-up
    char* ws = (char*)d_ws;
    auto aup = [](size_t v) { return (v + 255) & ~(size_t)255; };
    size_t off = 0;
    unsigned* wpack   = (unsigned*)(ws + off); off = aup(off + (size_t)WPACK_DW * 4);
    float*    xtail   = (float*)   (ws + off); off = aup(off + (size_t)N_NODES * 32 * 4);
    float*    nodecat = (float*)   (ws + off); off = aup(off + (size_t)N_NODES * NCAT * 4);
    float*    agg     = (float*)   (ws + off); off = aup(off + (size_t)N_NODES * 8 * 4);
    float*    hdot    = (float*)   (ws + off); off = aup(off + (size_t)N_NODES * 4);
    unsigned* smaxkey = (unsigned*)(ws + off); off = aup(off + (size_t)N_NODES * 4);
    float*    denom   = (float*)   (ws + off); off = aup(off + (size_t)N_NODES * 4);
    float*    hl2     = (float*)   (ws + off); off = aup(off + (size_t)N_NODES * 8 * 4);
    float*    escore  = (float*)   (ws + off); off = aup(off + (size_t)N_EDGES * 4);

    const int NB  = (N_NODES + 255) / 256;            // 391
    const int EB  = (N_EDGES + 255) / 256;            // 12500
    const int GB  = (NTILES + 7) / 8;                 // 782
    const int PB  = (WPACK_DW + 255) / 256;           // 90
    const int TB  = (N_NODES * 32 + 255) / 256;       // 12500

    // ---------------- layer 1 ----------------
    pack_w1_kernel<<<PB, 256, 0, stream>>>(w1, root1, wpack);
    pack_xtail_kernel<<<TB, 256, 0, stream>>>(x, xtail);
    gemm1_wmma_kernel<<<GB, 256, WLDS_BYTES, stream>>>(x, xtail, wpack, nodecat);
    node_hdot_init_kernel<8><<<NB, 256, 0, stream>>>(nodecat, att1, hdot, smaxkey, denom, agg);
    edge_score_max_kernel<8><<<EB, 256, 0, stream>>>(ei, pseudo, nodecat, hdot, att1, escore, smaxkey);
    edge_expsum_kernel<<<EB, 256, 0, stream>>>(ei, escore, smaxkey, denom);
    edge_agg_kernel<8><<<EB, 256, 0, stream>>>(ei, pseudo, nodecat, escore, denom, agg);
    node_out1_kernel<<<NB, 256, 0, stream>>>(nodecat, agg, bias1, hl2);

    // ---------------- layer 2 ----------------
    node_l2_kernel<<<NB, 256, 0, stream>>>(hl2, w2, root2, nodecat);
    node_hdot_init_kernel<7><<<NB, 256, 0, stream>>>(nodecat, att2, hdot, smaxkey, denom, agg);
    edge_score_max_kernel<7><<<EB, 256, 0, stream>>>(ei, pseudo, nodecat, hdot, att2, escore, smaxkey);
    edge_expsum_kernel<<<EB, 256, 0, stream>>>(ei, escore, smaxkey, denom);
    edge_agg_kernel<7><<<EB, 256, 0, stream>>>(ei, pseudo, nodecat, escore, denom, agg);
    node_final_kernel<<<NB, 256, 0, stream>>>(nodecat, agg, bias2, out);
}
